// SpatialBasicBlock_34763465294370
// MI455X (gfx1250) — matrix operand order repaired and checked by reading, not compiled, not run on hardware
//
#include <hip/hip_runtime.h>

typedef _Float16 f16;
typedef __attribute__((ext_vector_type(16))) _Float16 v16h;
typedef __attribute__((ext_vector_type(8)))  _Float16 v8h;
typedef __attribute__((ext_vector_type(8)))  float    v8f;

#define Nn 64
#define Cc 64
#define Tt 512
#define Vv 25
#define Kk 3
#define TB 16            // timesteps per workgroup
#define P  (TB*Vv)       // 400 columns per tile (exact multiple of 16)
#define XP 72            // Xlds pitch in halves (144B, 16B aligned)
#define WP 64            // Wlds pitch in halves (128B)
#define YP 96            // Ylds kv pitch (k*32+v, padded; 192B rows)
#define AP 96            // AeT pitch
#define NTHREADS 512

__launch_bounds__(NTHREADS, 1)
__global__ void stgcn_fused(const float* __restrict__ xg,
                            const float* __restrict__ Wg,
                            const float* __restrict__ bg,
                            const float* __restrict__ Ag,
                            const float* __restrict__ Eg,
                            const float* __restrict__ gg,
                            const float* __restrict__ betag,
                            const float* __restrict__ mg,
                            const float* __restrict__ vg,
                            float* __restrict__ outg)
{
    __shared__ __attribute__((aligned(16))) f16  Wlds[192 * WP];      // 24.0 KB
    __shared__ __attribute__((aligned(16))) f16  Xlds[P * XP];        // 56.25 KB, [p][c]
    __shared__ __attribute__((aligned(16))) f16  Ylds[TB * 64 * YP];  // 192 KB, [lt][c][k*32+v]
    __shared__ __attribute__((aligned(16))) f16  AeT[32 * AP];        // 6 KB, [w][k*32+v]
    __shared__ __attribute__((aligned(16))) float bias2[64 * 32];     // 8 KB
    __shared__ float invs[64];
    __shared__ float adds[64];

    const int tid = threadIdx.x;
    const int n   = blockIdx.y;
    const int t0  = blockIdx.x * TB;

    // ---------------- prologue: stage operands into LDS ----------------
    for (int e = tid; e < 192 * 64; e += NTHREADS)
        Wlds[(e >> 6) * WP + (e & 63)] = (f16)Wg[e];

    {   // x tile, transposed to [p][c] (global reads coalesced along p)
        const float* xb = xg + ((size_t)n * Cc) * Tt * Vv + (size_t)t0 * Vv;
        for (int e = tid; e < Cc * P; e += NTHREADS) {
            int c = e / P, p = e - c * P;
            Xlds[p * XP + c] = (f16)xb[(size_t)c * Tt * Vv + p];
        }
    }
    for (int e = tid; e < 32 * AP; e += NTHREADS) AeT[e] = (f16)0.f;
    __syncthreads();  // zeros visible before the scattered fill below
    for (int e = tid; e < Kk * Vv * Vv; e += NTHREADS) {
        int k = e / (Vv * Vv), r = e - k * Vv * Vv;
        int v = r / Vv, w = r - v * Vv;
        AeT[w * AP + k * 32 + v] = (f16)(Ag[e] * Eg[e]);
    }
    // bias2[c][w] = sum_k b[k*64+c] * sum_v Ae[k,v,w]
    for (int e = tid; e < 64 * 32; e += NTHREADS) {
        int c = e >> 5, w = e & 31;
        float s = 0.f;
        if (w < Vv) {
            for (int k = 0; k < Kk; ++k) {
                float cs = 0.f;
                const float* Ak = Ag + k * Vv * Vv;
                const float* Ek = Eg + k * Vv * Vv;
                for (int v = 0; v < Vv; ++v) cs += Ak[v * Vv + w] * Ek[v * Vv + w];
                s += bg[k * 64 + c] * cs;
            }
        }
        bias2[e] = s;
    }
    if (tid < 64) {
        float iv = gg[tid] * rsqrtf(vg[tid] + 1e-5f);
        invs[tid] = iv;
        adds[tid] = betag[tid] - mg[tid] * iv;
    }
    __syncthreads();

    const int lane = tid & 31;
    // Scalar (SGPR) wave id: forces the job loops below into *uniform* scalar
    // control flow, guaranteeing EXEC == all-1s at every WMMA (ISA §7.12).
    const int wave = __builtin_amdgcn_readfirstlane(tid >> 5);   // 0..15
    const int l    = lane & 15;
    const int hid  = lane >> 4;      // half-wave id

    // ------------- stage 1: Y[o][p] = W[o][c] * X[c][p], K=64 -------------
    for (int job = wave; job < 12 * 25; job += 16) {
        int ot = job / 25, pt = job - ot * 25;
        v8f acc = {};
#pragma unroll
        for (int kc = 0; kc < 2; ++kc) {
            v16h a, bm;
            // A (16x32 f16): lane<16 row o, K = kc*32 + {0..7, 16..23}; lane>=16 +8
            const f16* ap = &Wlds[(ot * 16 + l) * WP + kc * 32 + hid * 8];
            ((v8h*)&a)[0] = *(const v8h*)ap;
            ((v8h*)&a)[1] = *(const v8h*)(ap + 16);
            // B (32x16 f16): lane<16 col p, K = kc*32 + {0..7, 16..23}; lane>=16 +8
            const f16* bp = &Xlds[(pt * 16 + l) * XP + kc * 32 + hid * 8];
            ((v8h*)&bm)[0] = *(const v8h*)bp;
            ((v8h*)&bm)[1] = *(const v8h*)(bp + 16);
            acc = __builtin_amdgcn_wmma_f32_16x16x32_f16(false, a, false, bm,
                                                         (short)0, acc, false, false);
        }
        // D: lane holds column p = pt*16+l ; rows o = ot*16 + hid*8 + r
        int p  = pt * 16 + l;
        int lt = p / Vv;
        int v  = p - lt * Vv;
#pragma unroll
        for (int r = 0; r < 8; ++r) {
            int o = ot * 16 + hid * 8 + r;
            int k = o >> 6, c = o & 63;
            Ylds[(lt * 64 + c) * YP + k * 32 + v] = (f16)acc[r];
        }
    }
    __syncthreads();

    // --------- stage 2: out[c][w] = Y_lt[c][kv] * AeT[kv][w], K=96 ---------
    const float* xres = xg  + (((size_t)n * Cc) * Tt + t0) * Vv;
    float*       ob   = outg + (((size_t)n * Cc) * Tt + t0) * Vv;
    for (int job = wave; job < TB * 4 * 2; job += 16) {
        int lt = job >> 3;
        int ct = (job >> 1) & 3;
        int wt = job & 1;
        v8f acc = {};
#pragma unroll
        for (int kc = 0; kc < 3; ++kc) {
            v16h a, bm;
            const f16* ap = &Ylds[(lt * 64 + ct * 16 + l) * YP + kc * 32 + hid * 8];
            ((v8h*)&a)[0] = *(const v8h*)ap;
            ((v8h*)&a)[1] = *(const v8h*)(ap + 16);
            const f16* bp = &AeT[(wt * 16 + l) * AP + kc * 32 + hid * 8];
            ((v8h*)&bm)[0] = *(const v8h*)bp;
            ((v8h*)&bm)[1] = *(const v8h*)(bp + 16);
            acc = __builtin_amdgcn_wmma_f32_16x16x32_f16(false, a, false, bm,
                                                         (short)0, acc, false, false);
        }
        // epilogue: bias2 + BN + residual + relu (masked; no WMMA below)
        int w = wt * 16 + l;
        int t = lt;
        if (w < Vv) {
#pragma unroll
            for (int r = 0; r < 8; ++r) {
                int c = ct * 16 + hid * 8 + r;
                float val = (acc[r] + bias2[(c << 5) + w]) * invs[c] + adds[c];
                val += xres[((size_t)c * Tt + t) * Vv + w];
                ob[((size_t)c * Tt + t) * Vv + w] = fmaxf(val, 0.f);
            }
        }
    }
}

extern "C" void kernel_launch(void* const* d_in, const int* in_sizes, int n_in,
                              void* d_out, int out_size, void* d_ws, size_t ws_size,
                              hipStream_t stream) {
    (void)in_sizes; (void)n_in; (void)out_size; (void)d_ws; (void)ws_size;
    dim3 grid(Tt / TB, Nn);   // 32 x 64 = 2048 workgroups
    stgcn_fused<<<grid, NTHREADS, 0, stream>>>(
        (const float*)d_in[0],  // x
        (const float*)d_in[1],  // W
        (const float*)d_in[2],  // b
        (const float*)d_in[3],  // A
        (const float*)d_in[4],  // edge
        (const float*)d_in[5],  // gamma
        (const float*)d_in[6],  // beta
        (const float*)d_in[7],  // mean
        (const float*)d_in[8],  // var
        (float*)d_out);
}
